// MultiheadAttention_43293270344025
// MI455X (gfx1250) — compile-verified, matrix operand order checked by reading, not run on hardware
//
#include <hip/hip_runtime.h>
#include <hip/hip_bf16.h>

// MI455X / gfx1250, wave32, WMMA f16 path.
// Shapes: B=4, T=2048, C=1024, H=16, D=64, D_MODEL=1024, FF=4096.

typedef __attribute__((ext_vector_type(16))) _Float16 v16h;
typedef __attribute__((ext_vector_type(8)))  _Float16 h8v;
typedef __attribute__((ext_vector_type(8)))  float    v8f;

#define BATCH 4
#define CTX   2048
#define NEMBD 1024
#define NHEAD 16
#define HSZ   64
#define DMODEL 1024
#define DFF    4096
#define MTOT  (BATCH*CTX)          // 8192

// ---------------------------------------------------------------- conversions

__global__ __launch_bounds__(256) void k_cvt_f32_f16(const float* __restrict__ s,
                                                     _Float16* __restrict__ d, int n) {
    int i = blockIdx.x * 256 + threadIdx.x;
    if (i < n) d[i] = (_Float16)s[i];
}

// Wq/Wk/Wv: [H=16][C=1024][D=64] f32 -> Wt [N=H*D=1024][K=C=1024] f16
// (B operand pre-transposed so GEMM stages it with contiguous b128 strips)
__global__ __launch_bounds__(256) void k_cvt_wqkv(const float* __restrict__ W,
                                                  _Float16* __restrict__ Wt) {
    int i = blockIdx.x * 256 + threadIdx.x;          // 0 .. 1M-1
    int d = i & 63;
    int c = (i >> 6) & 1023;
    int h = i >> 16;
    Wt[(size_t)(h * 64 + d) * 1024 + c] = (_Float16)W[i];
}

// W [Kd][Nd] f32 -> Wt [Nd][Kd] f16
__global__ __launch_bounds__(256) void k_cvt_transpose(const float* __restrict__ W,
                                                       _Float16* __restrict__ Wt,
                                                       int Kd, int Nd) {
    int i = blockIdx.x * 256 + threadIdx.x;
    if (i >= Kd * Nd) return;
    int n = i % Nd;
    int k = i / Nd;
    Wt[(size_t)n * Kd + k] = (_Float16)W[i];
}

// V [B*T][H*D] f16 -> Vt [(b*H+h)*64 + d][T] f16  (contiguous along t)
__global__ __launch_bounds__(256) void k_transpose_v(const _Float16* __restrict__ V,
                                                     _Float16* __restrict__ Vt) {
    int i = blockIdx.x * 256 + threadIdx.x;          // 0 .. 8M-1
    int d = i & 63;
    int h = (i >> 6) & 15;
    int row = i >> 10;                                // b*T + t
    int t = row & 2047;
    int b = row >> 11;
    Vt[((size_t)((b * 16 + h) * 64 + d)) * 2048 + t] = V[(size_t)i];
}

// ---------------------------------------------------------------- generic GEMM
// C[M,N] = A[M,K] * Bt[N,K]^T (+bias) (relu?)   A,Bt f16 row-major.
// Block: 256 thr = 8 waves; tile 128x128; K-step 64; register-prefetch double
// buffering in LDS (one barrier per K-tile). M,N mult of 128; K mult of 64.

#define LSTR 72   // LDS row stride in halves (64 + 8 pad); 144B = 16B aligned

template <bool OUT_F16, bool RELU>
__global__ __launch_bounds__(256) void k_gemm_f16(const _Float16* __restrict__ A,
                                                  const _Float16* __restrict__ Bt,
                                                  const float* __restrict__ bias,
                                                  void* __restrict__ Cout,
                                                  int M, int N, int K) {
    __shared__ alignas(16) _Float16 As[2 * 128 * LSTR];
    __shared__ alignas(16) _Float16 Bs[2 * 128 * LSTR];

    const int tid  = threadIdx.x;
    const int lane = tid & 31;
    const int wave = tid >> 5;
    const int l16  = lane & 15;
    const int grp  = lane >> 4;
    const int n0   = blockIdx.x * 128;
    const int m0   = blockIdx.y * 128;

    // strip decomposition: 128 rows x 64 halves = 1024 strips of 8; 4/thread
    const int srow[4] = { (tid + 0) >> 3, (tid + 256) >> 3, (tid + 512) >> 3, (tid + 768) >> 3 };
    const int skc     = (tid & 7) * 8;

    v8f z = {};
    v8f acc[8];
#pragma unroll
    for (int j = 0; j < 8; ++j) acc[j] = z;

    const int ntiles = K >> 6;

    // stage tile 0 into buffer 0
#pragma unroll
    for (int i = 0; i < 4; ++i) {
        h8v va = *(const h8v*)(A  + (size_t)(m0 + srow[i]) * K + skc);
        h8v vb = *(const h8v*)(Bt + (size_t)(n0 + srow[i]) * K + skc);
        *(h8v*)(As + srow[i] * LSTR + skc) = va;
        *(h8v*)(Bs + srow[i] * LSTR + skc) = vb;
    }
    __syncthreads();

    for (int t = 0; t < ntiles; ++t) {
        const int buf  = (t & 1) * 128 * LSTR;
        const int nbuf = ((t + 1) & 1) * 128 * LSTR;
        const bool more = (t + 1) < ntiles;

        // prefetch next K-tile into registers (hides under WMMA stream)
        h8v pa[4], pb[4];
        if (more) {
            const int kt = (t + 1) << 6;
#pragma unroll
            for (int i = 0; i < 4; ++i) {
                pa[i] = *(const h8v*)(A  + (size_t)(m0 + srow[i]) * K + kt + skc);
                pb[i] = *(const h8v*)(Bt + (size_t)(n0 + srow[i]) * K + kt + skc);
            }
        }

        // compute: two 32-deep chunks, 8 N-tiles each
#pragma unroll
        for (int c = 0; c < 2; ++c) {
            v16h a;
            {
                const _Float16* ap = As + buf + (16 * wave + l16) * LSTR + 32 * c + 8 * grp;
                h8v lo = *(const h8v*)ap;
                h8v hi = *(const h8v*)(ap + 16);
#pragma unroll
                for (int i = 0; i < 8; ++i) { a[i] = lo[i]; a[8 + i] = hi[i]; }
            }
#pragma unroll
            for (int j = 0; j < 8; ++j) {
                v16h b;
                const _Float16* bp = Bs + buf + (16 * j + l16) * LSTR + 32 * c + 8 * grp;
                h8v lo = *(const h8v*)bp;
                h8v hi = *(const h8v*)(bp + 16);
#pragma unroll
                for (int i = 0; i < 8; ++i) { b[i] = lo[i]; b[8 + i] = hi[i]; }
                acc[j] = __builtin_amdgcn_wmma_f32_16x16x32_f16(
                    false, a, false, b, (short)0, acc[j], false, false);
            }
        }

        if (more) {
#pragma unroll
            for (int i = 0; i < 4; ++i) {
                *(h8v*)(As + nbuf + srow[i] * LSTR + skc) = pa[i];
                *(h8v*)(Bs + nbuf + srow[i] * LSTR + skc) = pb[i];
            }
        }
        __syncthreads();
    }

    // epilogue (C/D layout: VGPR r -> row r+8*grp; lane%16 -> col)
#pragma unroll
    for (int j = 0; j < 8; ++j) {
#pragma unroll
        for (int r = 0; r < 8; ++r) {
            int row = m0 + 16 * wave + r + 8 * grp;
            int col = n0 + 16 * j + l16;
            float v = acc[j][r];
            if (bias) v += bias[col];
            if (RELU) v = v > 0.f ? v : 0.f;
            if (OUT_F16) ((_Float16*)Cout)[(size_t)row * N + col] = (_Float16)v;
            else         ((float*)Cout)[(size_t)row * N + col] = v;
        }
    }
}

// ---------------------------------------------------------------- attention
// Flash-attention: grid (T/128, H, B); 8 waves; each wave = one 16-row Q tile.
// Q,K: [B*T][H*64] f16. Vt: [(b*H+h)*64+d][T] f16. Att out: [B*T][H*64] f16.

__global__ __launch_bounds__(256) void k_attn(const _Float16* __restrict__ Q,
                                              const _Float16* __restrict__ Km,
                                              const _Float16* __restrict__ Vt,
                                              _Float16* __restrict__ Att) {
    __shared__ alignas(16) _Float16 Plds[8 * 16 * 40];

    const int tid  = threadIdx.x;
    const int lane = tid & 31;
    const int wave = tid >> 5;
    const int l16  = lane & 15;
    const int grp  = lane >> 4;
    const int h = blockIdx.y, b = blockIdx.z;
    const int t0 = blockIdx.x * 128 + wave * 16;
    _Float16* P = Plds + wave * 16 * 40;

    const size_t rowbase = (size_t)(b * CTX) * DMODEL + h * 64;  // + t*1024
    const size_t vtb = (size_t)((b * NHEAD + h) * 64) * CTX;     // + d*2048 + t

    // Q A-fragments (K-dim 64 split into two 32-chunks)
    v16h qa[2];
#pragma unroll
    for (int c = 0; c < 2; ++c) {
        const _Float16* p = Q + rowbase + (size_t)(t0 + l16) * DMODEL + 32 * c + 8 * grp;
        h8v lo = *(const h8v*)p;
        h8v hi = *(const h8v*)(p + 16);
#pragma unroll
        for (int i = 0; i < 8; ++i) { qa[c][i] = lo[i]; qa[c][8 + i] = hi[i]; }
    }

    v8f z = {};
    v8f o[4];
#pragma unroll
    for (int j = 0; j < 4; ++j) o[j] = z;
    float rm[8], rl[8];
#pragma unroll
    for (int r = 0; r < 8; ++r) { rm[r] = -1e30f; rl[r] = 0.f; }
    const float scale = 0.125f;  // 1/sqrt(64)

    for (int ks = 0; ks <= t0 + 15; ks += 32) {   // causal: only key blocks <= t0+15
        // S = Q * K^T, two 16-key tiles
        v8f s0 = z, s1 = z;
#pragma unroll
        for (int c = 0; c < 2; ++c) {
            v16h kb0, kb1;
            const _Float16* p0 = Km + rowbase + (size_t)(ks + l16) * DMODEL + 32 * c + 8 * grp;
            const _Float16* p1 = p0 + (size_t)16 * DMODEL;
            h8v lo = *(const h8v*)p0, hi = *(const h8v*)(p0 + 16);
#pragma unroll
            for (int i = 0; i < 8; ++i) { kb0[i] = lo[i]; kb0[8 + i] = hi[i]; }
            lo = *(const h8v*)p1; hi = *(const h8v*)(p1 + 16);
#pragma unroll
            for (int i = 0; i < 8; ++i) { kb1[i] = lo[i]; kb1[8 + i] = hi[i]; }
            s0 = __builtin_amdgcn_wmma_f32_16x16x32_f16(false, qa[c], false, kb0, (short)0, s0, false, false);
            s1 = __builtin_amdgcn_wmma_f32_16x16x32_f16(false, qa[c], false, kb1, (short)0, s1, false, false);
        }

        // online softmax (rows live in 16-lane groups; butterfly reduce)
        float p0v[8], p1v[8];
#pragma unroll
        for (int r = 0; r < 8; ++r) {
            int trow = t0 + r + 8 * grp;
            int k0 = ks + l16, k1 = k0 + 16;
            float a0 = (k0 <= trow) ? s0[r] * scale : -1e30f;
            float a1 = (k1 <= trow) ? s1[r] * scale : -1e30f;
            float mx = fmaxf(a0, a1);
            for (int dd = 1; dd < 16; dd <<= 1) mx = fmaxf(mx, __shfl_xor(mx, dd, 32));
            float mnew = fmaxf(rm[r], mx);
            float e0 = __expf(a0 - mnew), e1 = __expf(a1 - mnew);
            float sum = e0 + e1;
            for (int dd = 1; dd < 16; dd <<= 1) sum += __shfl_xor(sum, dd, 32);
            float f = __expf(rm[r] - mnew);
            rl[r] = rl[r] * f + sum;
            rm[r] = mnew;
            p0v[r] = e0; p1v[r] = e1;
#pragma unroll
            for (int j = 0; j < 4; ++j) o[j][r] *= f;
        }

        // P: D-layout -> LDS -> A-layout (same-wave DS is in-order; fence compiler)
        asm volatile("s_wait_dscnt 0" ::: "memory");
#pragma unroll
        for (int r = 0; r < 8; ++r) {
            P[(r + 8 * grp) * 40 + l16]      = (_Float16)p0v[r];
            P[(r + 8 * grp) * 40 + 16 + l16] = (_Float16)p1v[r];
        }
        asm volatile("s_wait_dscnt 0" ::: "memory");
        v16h pa;
        {
            h8v lo = *(const h8v*)(P + l16 * 40 + 8 * grp);
            h8v hi = *(const h8v*)(P + l16 * 40 + 16 + 8 * grp);
#pragma unroll
            for (int i = 0; i < 8; ++i) { pa[i] = lo[i]; pa[8 + i] = hi[i]; }
        }

        // O += P * V   (V^T rows contiguous along t -> b128 loads)
#pragma unroll
        for (int j = 0; j < 4; ++j) {
            v16h vb;
            const _Float16* p = Vt + vtb + (size_t)(16 * j + l16) * CTX + ks + 8 * grp;
            h8v lo = *(const h8v*)p, hi = *(const h8v*)(p + 16);
#pragma unroll
            for (int i = 0; i < 8; ++i) { vb[i] = lo[i]; vb[8 + i] = hi[i]; }
            o[j] = __builtin_amdgcn_wmma_f32_16x16x32_f16(false, pa, false, vb, (short)0, o[j], false, false);
        }
    }

    // normalize + store concat-head output
#pragma unroll
    for (int r = 0; r < 8; ++r) {
        float inv = 1.f / rl[r];
        int trow = t0 + r + 8 * grp;
#pragma unroll
        for (int j = 0; j < 4; ++j) {
            Att[(size_t)(b * CTX + trow) * DMODEL + h * 64 + 16 * j + l16] =
                (_Float16)(o[j][r] * inv);
        }
    }
}

// ---------------------------------------------------------------- launch

extern "C" void kernel_launch(void* const* d_in, const int* in_sizes, int n_in,
                              void* d_out, int out_size, void* d_ws, size_t ws_size,
                              hipStream_t stream) {
    (void)in_sizes; (void)n_in; (void)out_size; (void)ws_size;

    const float* x  = (const float*)d_in[0];
    const float* Wq = (const float*)d_in[1];
    const float* Wk = (const float*)d_in[2];
    const float* Wv = (const float*)d_in[3];
    const float* W1 = (const float*)d_in[4];
    const float* b1 = (const float*)d_in[5];
    const float* W2 = (const float*)d_in[6];
    const float* b2 = (const float*)d_in[7];
    float* out = (float*)d_out;

    // workspace carve-up (f16 elements)
    _Float16* ws  = (_Float16*)d_ws;
    _Float16* xh  = ws;                              // 8192*1024
    _Float16* wqt = xh  + (size_t)MTOT * NEMBD;      // [1024n][1024k]
    _Float16* wkt = wqt + (size_t)NEMBD * DMODEL;
    _Float16* wvt = wkt + (size_t)NEMBD * DMODEL;
    _Float16* qh  = wvt + (size_t)NEMBD * DMODEL;    // 8192*1024
    _Float16* kh  = qh  + (size_t)MTOT * DMODEL;
    _Float16* vh  = kh  + (size_t)MTOT * DMODEL;
    _Float16* vth = vh  + (size_t)MTOT * DMODEL;
    _Float16* att = vth + (size_t)MTOT * DMODEL;
    _Float16* w1t = att + (size_t)MTOT * DMODEL;     // [4096n][1024k]
    _Float16* w2t = w1t + (size_t)DMODEL * DFF;      // [1024n][4096k]
    _Float16* h1  = w2t + (size_t)DFF * DMODEL;      // 8192*4096

    const int nX = MTOT * NEMBD;                     // 8388608
    k_cvt_f32_f16<<<nX / 256, 256, 0, stream>>>(x, xh, nX);
    k_cvt_wqkv<<<(NHEAD * NEMBD * HSZ) / 256, 256, 0, stream>>>(Wq, wqt);
    k_cvt_wqkv<<<(NHEAD * NEMBD * HSZ) / 256, 256, 0, stream>>>(Wk, wkt);
    k_cvt_wqkv<<<(NHEAD * NEMBD * HSZ) / 256, 256, 0, stream>>>(Wv, wvt);
    k_cvt_transpose<<<(DMODEL * DFF) / 256, 256, 0, stream>>>(W1, w1t, DMODEL, DFF);
    k_cvt_transpose<<<(DFF * DMODEL) / 256, 256, 0, stream>>>(W2, w2t, DFF, DMODEL);

    // Q/K/V projections: [8192,1024] x [1024,1024]^T -> f16
    dim3 gp(DMODEL / 128, MTOT / 128);
    k_gemm_f16<true, false><<<gp, 256, 0, stream>>>(xh, wqt, nullptr, qh, MTOT, DMODEL, NEMBD);
    k_gemm_f16<true, false><<<gp, 256, 0, stream>>>(xh, wkt, nullptr, kh, MTOT, DMODEL, NEMBD);
    k_gemm_f16<true, false><<<gp, 256, 0, stream>>>(xh, wvt, nullptr, vh, MTOT, DMODEL, NEMBD);

    k_transpose_v<<<nX / 256, 256, 0, stream>>>(vh, vth);

    // causal flash attention -> att [8192,1024] f16
    k_attn<<<dim3(CTX / 128, NHEAD, BATCH), 256, 0, stream>>>(qh, kh, vth, att);

    // FFN1: [8192,1024] x [4096,1024]^T + b1 -> f16
    k_gemm_f16<true, false><<<dim3(DFF / 128, MTOT / 128), 256, 0, stream>>>(
        att, w1t, b1, h1, MTOT, DFF, DMODEL);
    // FFN2: [8192,4096] x [1024,4096]^T + b2, ReLU -> f32 output
    k_gemm_f16<false, true><<<dim3(DMODEL / 128, MTOT / 128), 256, 0, stream>>>(
        h1, w2t, b2, out, MTOT, DMODEL, DFF);
}
